// NER_LSTMNet_29137058136579
// MI455X (gfx1250) — matrix operand order repaired and checked by reading, not compiled, not run on hardware
//
#include <hip/hip_runtime.h>
#include <hip/hip_bf16.h>

typedef __attribute__((ext_vector_type(16))) _Float16 v16h;
typedef __attribute__((ext_vector_type(8)))  _Float16 v8h;
typedef __attribute__((ext_vector_type(8)))  float    v8f;

#define DEV __device__ __forceinline__

DEV float sigmoidf_(float x) { return 1.0f / (1.0f + __expf(-x)); }

// A-fragment (16x32 f16, row-major source, lda in elements).
// ISA layout: lane l<16 -> row m=l, K groups {0..7}+{16..23}; lane l>=16 -> K {8..15}+{24..31}.
DEV v16h load_frag_a(const _Float16* __restrict__ base, int row, int lda, int kBase) {
  const int lane = threadIdx.x & 31;
  const _Float16* p = base + (size_t)(row + (lane & 15)) * lda + (size_t)kBase + ((lane >> 4) << 3);
  union { v16h v; v8h h[2]; } u;
  u.h[0] = *(const v8h*)p;
  u.h[1] = *(const v8h*)(p + 16);
  return u.v;
}

// B-fragment (32x16 f16) loaded from row-major W[N][K] (i.e. column-major B).
// ISA layout: lane l<16 -> column n=l, K groups {0..7}+{16..23}; lane l>=16 -> K {8..15}+{24..31}.
DEV v16h load_frag_b(const _Float16* __restrict__ base, int nRow, int ldb, int kBase) {
  const int lane = threadIdx.x & 31;
  const _Float16* p = base + (size_t)(nRow + (lane & 15)) * ldb + (size_t)kBase + ((lane >> 4) << 3);
  union { v16h v; v8h h[2]; } u;
  u.h[0] = *(const v8h*)p;
  u.h[1] = *(const v8h*)(p + 16);
  return u.v;
}

DEV v8f wmma16(v16h a, v16h b, v8f c) {
  return __builtin_amdgcn_wmma_f32_16x16x32_f16(false, a, false, b, (short)0, c, false, false);
}

// ---------------------------------------------------------------------------
// Fused LSTM cell step: G = [A1 | Hprev] @ Wc^T + bias ; gate math ; update C, Hnext.
// Grid: 32 blocks x 256 threads. Block b owns columns [16b,16b+16) of each gate.
// A1: [256][K1] f16 (row stride = K1). Hprev: [256][512] f16. Wc: [2048][K1+512] f16.
// ---------------------------------------------------------------------------
__global__ __launch_bounds__(256, 1)
void lstm_cell_step(const _Float16* __restrict__ A1, int K1,
                    const _Float16* __restrict__ Hprev,
                    const _Float16* __restrict__ Wc, int Kc,
                    const float* __restrict__ bias,
                    float* __restrict__ Cst,
                    _Float16* __restrict__ Hnext) {
  const int wave = threadIdx.x >> 5;
  const int lane = threadIdx.x & 31;
  const int cb = blockIdx.x << 4;   // column base inside each gate block (0..496)
  const int mb = wave << 5;         // rows mb..mb+31 for this wave

  v8f acc[4][2] = {};

  for (int kb = 0; kb < K1; kb += 32) {
    v16h a0 = load_frag_a(A1, mb,      K1, kb);
    v16h a1 = load_frag_a(A1, mb + 16, K1, kb);
#pragma unroll
    for (int g = 0; g < 4; ++g) {
      v16h b = load_frag_b(Wc, (g << 9) + cb, Kc, kb);
      acc[g][0] = wmma16(a0, b, acc[g][0]);
      acc[g][1] = wmma16(a1, b, acc[g][1]);
    }
  }
  for (int kb = 0; kb < 512; kb += 32) {
    v16h a0 = load_frag_a(Hprev, mb,      512, kb);
    v16h a1 = load_frag_a(Hprev, mb + 16, 512, kb);
#pragma unroll
    for (int g = 0; g < 4; ++g) {
      v16h b = load_frag_b(Wc, (g << 9) + cb, Kc, K1 + kb);
      acc[g][0] = wmma16(a0, b, acc[g][0]);
      acc[g][1] = wmma16(a1, b, acc[g][1]);
    }
  }

  const int col = cb + (lane & 15);
  const int rhi = (lane >> 4) << 3;
  const float bi = bias[col];
  const float bf = bias[512 + col];
  const float bg = bias[1024 + col];
  const float bo = bias[1536 + col];
#pragma unroll
  for (int t = 0; t < 2; ++t) {
#pragma unroll
    for (int r = 0; r < 8; ++r) {
      const int row = mb + (t << 4) + rhi + r;
      const float gi = acc[0][t][r] + bi;
      const float gf = acc[1][t][r] + bf;
      const float gg = acc[2][t][r] + bg;
      const float go = acc[3][t][r] + bo;
      const size_t idx = (size_t)row * 512 + col;
      const float cOld = Cst[idx];
      const float cNew = sigmoidf_(gf) * cOld + sigmoidf_(gi) * tanhf(gg);
      Cst[idx] = cNew;
      Hnext[idx] = (_Float16)(sigmoidf_(go) * tanhf(cNew));
    }
  }
}

// ---------------------------------------------------------------------------
// Decoder output step: logits = H1 @ Wf^T + b -> softmax (write probs) -> argmax
// -> one-hot feedback xin. One workgroup, 512 threads (16 waves).
// ---------------------------------------------------------------------------
__global__ __launch_bounds__(512, 1)
void dec_out_step(const _Float16* __restrict__ H1,
                  const _Float16* __restrict__ Wf,   // [32][512] f16, this step
                  const float* __restrict__ bfc,     // [32], this step
                  float* __restrict__ outp,          // [256][128][32] f32
                  _Float16* __restrict__ xin,        // [256][32] f16
                  int s) {
  __shared__ float logits[256 * 32];
  const int wave = threadIdx.x >> 5;
  const int lane = threadIdx.x & 31;
  const int mb = wave << 4;

  v8f a0 = {}, a1 = {};
  for (int kb = 0; kb < 512; kb += 32) {
    v16h av = load_frag_a(H1, mb, 512, kb);
    v16h b0 = load_frag_b(Wf, 0, 512, kb);
    v16h b1 = load_frag_b(Wf, 16, 512, kb);
    a0 = wmma16(av, b0, a0);
    a1 = wmma16(av, b1, a1);
  }
  const int col = lane & 15;
  const int rhi = (lane >> 4) << 3;
#pragma unroll
  for (int r = 0; r < 8; ++r) {
    const int row = mb + rhi + r;
    logits[row * 32 + col]      = a0[r] + bfc[col];
    logits[row * 32 + 16 + col] = a1[r] + bfc[16 + col];
  }
  __syncthreads();

  const int tid = threadIdx.x;
  if (tid < 256) {
    float mx = -3.4e38f;
    int am = 0;
#pragma unroll
    for (int o = 0; o < 32; ++o) {
      const float v = logits[tid * 32 + o];
      if (v > mx) { mx = v; am = o; }
    }
    float e[32];
    float sum = 0.0f;
#pragma unroll
    for (int o = 0; o < 32; ++o) {
      e[o] = __expf(logits[tid * 32 + o] - mx);
      sum += e[o];
    }
    const float inv = 1.0f / sum;
    float* orow = outp + ((size_t)tid * 128 + s) * 32;
#pragma unroll
    for (int o = 0; o < 32; ++o) {
      orow[o] = e[o] * inv;
      xin[tid * 32 + o] = (_Float16)(o == am ? 1.0f : 0.0f);
    }
  }
}

// ---------------------------------------------------------------------------
// Prep kernels (run each call; deterministic)
// ---------------------------------------------------------------------------
__global__ void k_embed(const int* __restrict__ x, const float* __restrict__ emb,
                        _Float16* __restrict__ xe) {
  const size_t total = (size_t)128 * 256 * 256;   // [S][B][E]
  const size_t stride = (size_t)gridDim.x * blockDim.x;
  for (size_t i = (size_t)blockIdx.x * blockDim.x + threadIdx.x; i < total; i += stride) {
    const int e = (int)(i & 255);
    const int b = (int)((i >> 8) & 255);
    const int s = (int)(i >> 16);
    const int tok = x[b * 128 + s];                 // x is [B][S]
    xe[i] = (_Float16)emb[(size_t)tok * 256 + e];
  }
}

__global__ void k_wcat(const float* __restrict__ W1, int K1,
                       const float* __restrict__ W2, int K2,
                       _Float16* __restrict__ dst) {
  const int Kc = K1 + K2;
  const size_t total = (size_t)2048 * Kc;
  const size_t stride = (size_t)gridDim.x * blockDim.x;
  for (size_t i = (size_t)blockIdx.x * blockDim.x + threadIdx.x; i < total; i += stride) {
    const int n = (int)(i / Kc);
    const int k = (int)(i % Kc);
    const float v = (k < K1) ? W1[(size_t)n * K1 + k] : W2[(size_t)n * K2 + (k - K1)];
    dst[i] = (_Float16)v;
  }
}

__global__ void k_cvt(const float* __restrict__ src, _Float16* __restrict__ dst, size_t n) {
  const size_t stride = (size_t)gridDim.x * blockDim.x;
  for (size_t i = (size_t)blockIdx.x * blockDim.x + threadIdx.x; i < n; i += stride)
    dst[i] = (_Float16)src[i];
}

__global__ void k_bias(const float* __restrict__ b1, const float* __restrict__ b2,
                       float* __restrict__ dst) {
  const int i = blockIdx.x * blockDim.x + threadIdx.x;
  if (i < 2048) dst[i] = b1[i] + b2[i];
}

__global__ void k_init(_Float16* h0, _Float16* h1, float* c0, float* c1, _Float16* xin) {
  const int i = blockIdx.x * blockDim.x + threadIdx.x;
  if (i < 256 * 512) { h0[i] = (_Float16)0.0f; h1[i] = (_Float16)0.0f; c0[i] = 0.0f; c1[i] = 0.0f; }
  if (i < 256 * 32)  { xin[i] = (_Float16)((i & 31) == 0 ? 1.0f : 0.0f); }
}

// ---------------------------------------------------------------------------
extern "C" void kernel_launch(void* const* d_in, const int* in_sizes, int n_in,
                              void* d_out, int out_size, void* d_ws, size_t ws_size,
                              hipStream_t stream) {
  (void)in_sizes; (void)n_in; (void)out_size; (void)ws_size;

  const int*   x     = (const int*)  d_in[0];
  const float* emb   = (const float*)d_in[1];
  const float* eWih0 = (const float*)d_in[2];
  const float* eWhh0 = (const float*)d_in[3];
  const float* ebih0 = (const float*)d_in[4];
  const float* ebhh0 = (const float*)d_in[5];
  const float* eWih1 = (const float*)d_in[6];
  const float* eWhh1 = (const float*)d_in[7];
  const float* ebih1 = (const float*)d_in[8];
  const float* ebhh1 = (const float*)d_in[9];
  const float* dWih0 = (const float*)d_in[10];
  const float* dWhh0 = (const float*)d_in[11];
  const float* dbih0 = (const float*)d_in[12];
  const float* dbhh0 = (const float*)d_in[13];
  const float* dWih1 = (const float*)d_in[14];
  const float* dWhh1 = (const float*)d_in[15];
  const float* dbih1 = (const float*)d_in[16];
  const float* dbhh1 = (const float*)d_in[17];
  const float* fcw   = (const float*)d_in[18];
  const float* fcb   = (const float*)d_in[19];
  float* out = (float*)d_out;

  char* ws = (char*)d_ws;
  size_t off = 0;
  auto take = [&](size_t bytes) -> char* {
    char* p = ws + off;
    off = (off + bytes + 255) & ~(size_t)255;
    return p;
  };

  _Float16* xe    = (_Float16*)take((size_t)128 * 256 * 256 * 2);  // [S][B][E]
  _Float16* W0e   = (_Float16*)take((size_t)2048 * 768 * 2);       // [Wih0|Whh0]
  _Float16* W1e   = (_Float16*)take((size_t)2048 * 1024 * 2);
  _Float16* W0d   = (_Float16*)take((size_t)2048 * 544 * 2);
  _Float16* W1d   = (_Float16*)take((size_t)2048 * 1024 * 2);
  _Float16* fcw16 = (_Float16*)take((size_t)128 * 32 * 512 * 2);
  float* B0e = (float*)take(2048 * 4);
  float* B1e = (float*)take(2048 * 4);
  float* B0d = (float*)take(2048 * 4);
  float* B1d = (float*)take(2048 * 4);
  _Float16* h0b[2] = { (_Float16*)take(256 * 512 * 2), (_Float16*)take(256 * 512 * 2) };
  _Float16* h1b[2] = { (_Float16*)take(256 * 512 * 2), (_Float16*)take(256 * 512 * 2) };
  float* c0 = (float*)take(256 * 512 * 4);
  float* c1 = (float*)take(256 * 512 * 4);
  _Float16* xin = (_Float16*)take(256 * 32 * 2);

  // --- prep ---
  k_embed<<<2048, 256, 0, stream>>>(x, emb, xe);
  k_wcat<<<1024, 256, 0, stream>>>(eWih0, 256, eWhh0, 512, W0e);
  k_wcat<<<1024, 256, 0, stream>>>(eWih1, 512, eWhh1, 512, W1e);
  k_wcat<<<1024, 256, 0, stream>>>(dWih0, 32,  dWhh0, 512, W0d);
  k_wcat<<<1024, 256, 0, stream>>>(dWih1, 512, dWhh1, 512, W1d);
  k_cvt<<<1024, 256, 0, stream>>>(fcw, fcw16, (size_t)128 * 32 * 512);
  k_bias<<<8, 256, 0, stream>>>(ebih0, ebhh0, B0e);
  k_bias<<<8, 256, 0, stream>>>(ebih1, ebhh1, B1e);
  k_bias<<<8, 256, 0, stream>>>(dbih0, dbhh0, B0d);
  k_bias<<<8, 256, 0, stream>>>(dbih1, dbhh1, B1d);
  k_init<<<512, 256, 0, stream>>>(h0b[0], h1b[0], c0, c1, xin);

  // --- encoder: 128 steps, two stacked cells per step ---
  for (int s = 0; s < 128; ++s) {
    const int p = s & 1;
    lstm_cell_step<<<32, 256, 0, stream>>>(xe + (size_t)s * 256 * 256, 256,
                                           h0b[p], W0e, 768, B0e, c0, h0b[p ^ 1]);
    lstm_cell_step<<<32, 256, 0, stream>>>(h0b[p ^ 1], 512,
                                           h1b[p], W1e, 1024, B1e, c1, h1b[p ^ 1]);
  }

  // --- decoder: carries encoder state; parity continues (128 is even) ---
  for (int t = 0; t < 128; ++t) {
    const int p = t & 1;
    lstm_cell_step<<<32, 256, 0, stream>>>(xin, 32,
                                           h0b[p], W0d, 544, B0d, c0, h0b[p ^ 1]);
    lstm_cell_step<<<32, 256, 0, stream>>>(h0b[p ^ 1], 512,
                                           h1b[p], W1d, 1024, B1d, c1, h1b[p ^ 1]);
    dec_out_step<<<1, 512, 0, stream>>>(h1b[p ^ 1], fcw16 + (size_t)t * 32 * 512,
                                        fcb + t * 32, out, xin, t);
  }
}
